// FlexSelfAttention_11699490914737
// MI455X (gfx1250) — matrix operand order repaired and checked by reading, not compiled, not run on hardware
//
#include <hip/hip_runtime.h>
#include <hip/hip_bf16.h>

typedef __bf16 bf16_t;
typedef __attribute__((ext_vector_type(16))) __bf16 v16bf;
typedef __attribute__((ext_vector_type(8)))  __bf16 v8bf;
typedef __attribute__((ext_vector_type(8)))  float  v8f;

#define T_SEQ  2048
#define DIM    2048
#define NHEADS 16
#define HDIM   128
#define FQKV   6144   // 3 * NHEADS * HDIM

// ---------------------------------------------------------------------------
// f32 -> bf16 elementwise convert
// ---------------------------------------------------------------------------
__global__ void f32_to_bf16_kernel(const float* __restrict__ in,
                                   bf16_t* __restrict__ out, int n) {
  int i = blockIdx.x * blockDim.x + threadIdx.x;
  if (i < n) out[i] = (bf16_t)in[i];
}

// ---------------------------------------------------------------------------
// C[M,N] (f32) = A[M,K] (bf16 row-major) * B[N,K]^T (bf16 row-major weights)
// block = 256 threads = 8 waves; each wave owns one 16x16 C tile.
// grid = (N/128, M/16)
// ---------------------------------------------------------------------------
__global__ void gemm_bf16_wmma(const bf16_t* __restrict__ A,
                               const bf16_t* __restrict__ B,
                               float* __restrict__ C, int M, int N, int K) {
  const int lane = threadIdx.x & 31;
  const int wave = threadIdx.x >> 5;
  const int l16  = lane & 15;
  const int hf   = lane >> 4;
  const int tile_n = (blockIdx.x * 8 + wave) * 16;
  const int tile_m = blockIdx.y * 16;

  const bf16_t* arow = A + (long)(tile_m + l16) * K;   // A: row M = l16
  const bf16_t* brow = B + (long)(tile_n + l16) * K;   // B: col N = l16

  v8f acc = {};
  for (int k0 = 0; k0 < K; k0 += 32) {
    __builtin_prefetch(arow + k0 + 512, 0, 0);
    __builtin_prefetch(brow + k0 + 512, 0, 0);
    // A frag: K = hf*16 + [0..15] contiguous
    v8bf alo = *(const v8bf*)(arow + k0 + hf * 16);
    v8bf ahi = *(const v8bf*)(arow + k0 + hf * 16 + 8);
    v16bf a;
#pragma unroll
    for (int i = 0; i < 8; ++i) { a[i] = alo[i]; a[i + 8] = ahi[i]; }
    // B frag: K = hf*16 + [0..15] contiguous
    v16bf b = *(const v16bf*)(brow + k0 + hf * 16);
    acc = __builtin_amdgcn_wmma_f32_16x16x32_bf16(false, a, false, b,
                                                  (short)0, acc, false, false);
  }
#pragma unroll
  for (int r = 0; r < 8; ++r) {
    int m = tile_m + hf * 8 + r;                       // D: row = r + 8*hf
    C[(long)m * N + tile_n + l16] = acc[r];
  }
}

// ---------------------------------------------------------------------------
// RMSNorm + rotary on q,k -> bf16 [T, H*D]; v -> transposed bf16 [H*D, T]
// grid = (H, T), block = 128 (one thread per d)
// ---------------------------------------------------------------------------
__global__ void qk_prep_kernel(const float* __restrict__ qkv,
                               bf16_t* __restrict__ qbf,
                               bf16_t* __restrict__ kbf,
                               bf16_t* __restrict__ vT) {
  const int h = blockIdx.x, t = blockIdx.y;
  const int d = threadIdx.x;
  const int lane = threadIdx.x & 31, w = threadIdx.x >> 5;

  const float* base = qkv + (long)t * FQKV + h * HDIM;
  float qv = base[d];
  float kv = base[DIM + d];
  float vv = base[2 * DIM + d];

  __shared__ float red[8];
  float qs = qv * qv, ks = kv * kv;
#pragma unroll
  for (int m = 16; m >= 1; m >>= 1) {
    qs += __shfl_xor(qs, m, 32);
    ks += __shfl_xor(ks, m, 32);
  }
  if (lane == 0) { red[w] = qs; red[4 + w] = ks; }
  __syncthreads();
  float qss = red[0] + red[1] + red[2] + red[3];
  float kss = red[4] + red[5] + red[6] + red[7];
  float qn = qv * rsqrtf(qss * (1.0f / HDIM) + 1e-6f);
  float kn = kv * rsqrtf(kss * (1.0f / HDIM) + 1e-6f);

  __shared__ float qsh[HDIM], ksh[HDIM];
  qsh[d] = qn; ksh[d] = kn;
  __syncthreads();

  bf16_t* qout = qbf + (long)t * DIM + h * HDIM;
  bf16_t* kout = kbf + (long)t * DIM + h * HDIM;
  if (d < HDIM / 2) {
    // angular_freq[j] = (1/1024)^(j/31) for j<32, else 0  (ln 1024 = 6.9314718)
    float af = (d < 32) ? expf(-(float)d * (6.93147181f / 31.0f)) : 0.0f;
    float th = (float)t * af;
    float c = cosf(th), s = sinf(th);
    float q1 = qsh[d], q2 = qsh[d + 64];
    float k1 = ksh[d], k2 = ksh[d + 64];
    qout[d]      = (bf16_t)( q1 * c + q2 * s);
    qout[d + 64] = (bf16_t)(-q1 * s + q2 * c);
    kout[d]      = (bf16_t)( k1 * c + k2 * s);
    kout[d + 64] = (bf16_t)(-k1 * s + k2 * c);
  }
  vT[(long)(h * HDIM + d) * T_SEQ + t] = (bf16_t)vv;
}

// ---------------------------------------------------------------------------
// Flash attention, causal. One wave per block; each wave owns 16 query rows
// of one head. KV processed in chunks of 32 (two 16x16 score tiles).
// grid = (T/16, H), block = 32.
// ---------------------------------------------------------------------------
__global__ void attn_kernel(const bf16_t* __restrict__ qbf,
                            const bf16_t* __restrict__ kbf,
                            const bf16_t* __restrict__ vT,
                            bf16_t* __restrict__ ybf) {
  const int qb = blockIdx.x, h = blockIdx.y;
  const int t0 = qb * 16;
  const int lane = threadIdx.x;
  const int l16 = lane & 15, hf = lane >> 4;
  const float scale = 0.08838834764831845f;   // 1/sqrt(128)

  __shared__ bf16_t Pl[16][32];               // probs tile, row m x col k

  // Q fragments: 4 chunks over D=128
  v16bf aq[4];
  const bf16_t* qrow = qbf + (long)(t0 + l16) * DIM + h * HDIM;
#pragma unroll
  for (int kc = 0; kc < 4; ++kc) {
    v8bf lo = *(const v8bf*)(qrow + kc * 32 + hf * 16);
    v8bf hi = *(const v8bf*)(qrow + kc * 32 + hf * 16 + 8);
#pragma unroll
    for (int i = 0; i < 8; ++i) { aq[kc][i] = lo[i]; aq[kc][i + 8] = hi[i]; }
  }

  float mrow[8], lrow[8];
  v8f accy[8];
#pragma unroll
  for (int r = 0; r < 8; ++r) {
    mrow[r] = -1e30f; lrow[r] = 0.0f;
#pragma unroll
    for (int j = 0; j < 8; ++j) accy[r][j] = 0.0f;
  }

  const int smax = t0 + 16;                   // exclusive KV bound (causal)
  for (int s0 = 0; s0 < smax; s0 += 32) {
    const bool has1 = (s0 + 16) < smax;       // wave-uniform

    // ---- S = Q K^T for two 16-col tiles ----
    v8f sa0 = {}, sa1 = {};
    const bf16_t* krow0 = kbf + (long)(s0 + l16) * DIM + h * HDIM;
#pragma unroll
    for (int kc = 0; kc < 4; ++kc) {
      v16bf b0 = *(const v16bf*)(krow0 + kc * 32 + hf * 16);
      sa0 = __builtin_amdgcn_wmma_f32_16x16x32_bf16(false, aq[kc], false, b0,
                                                    (short)0, sa0, false, false);
    }
    if (has1) {
      const bf16_t* krow1 = kbf + (long)(s0 + 16 + l16) * DIM + h * HDIM;
#pragma unroll
      for (int kc = 0; kc < 4; ++kc) {
        v16bf b1 = *(const v16bf*)(krow1 + kc * 32 + hf * 16);
        sa1 = __builtin_amdgcn_wmma_f32_16x16x32_bf16(false, aq[kc], false, b1,
                                                      (short)0, sa1, false, false);
      }
    }

    // ---- online softmax over this 32-col chunk ----
    float facs[8];
#pragma unroll
    for (int r = 0; r < 8; ++r) {
      int trow = t0 + hf * 8 + r;
      int sc0 = s0 + l16, sc1 = s0 + 16 + l16;
      float sv0 = (sc0 <= trow) ? sa0[r] * scale : -1e30f;
      float sv1 = (has1 && sc1 <= trow) ? sa1[r] * scale : -1e30f;
      float rmax = fmaxf(sv0, sv1);
#pragma unroll
      for (int m = 8; m >= 1; m >>= 1)
        rmax = fmaxf(rmax, __shfl_xor(rmax, m, 32));
      float mnew = fmaxf(mrow[r], rmax);
      float p0 = __expf(sv0 - mnew);
      float p1 = __expf(sv1 - mnew);
      float ps = p0 + p1;
#pragma unroll
      for (int m = 8; m >= 1; m >>= 1) ps += __shfl_xor(ps, m, 32);
      float fac = __expf(mrow[r] - mnew);
      lrow[r] = lrow[r] * fac + ps;
      mrow[r] = mnew;
      facs[r] = fac;
      Pl[hf * 8 + r][l16]      = (bf16_t)p0;   // D layout -> LDS
      Pl[hf * 8 + r][16 + l16] = (bf16_t)p1;
    }
#pragma unroll
    for (int tn = 0; tn < 8; ++tn)
#pragma unroll
      for (int r = 0; r < 8; ++r) accy[tn][r] *= facs[r];

    __syncthreads();   // single wave: S_NOP, but fences LDS ordering

    // ---- reload P in A-fragment layout, accumulate P V ----
    v16bf ap;
    v8bf plo = *(const v8bf*)&Pl[l16][hf * 16];
    v8bf phi = *(const v8bf*)&Pl[l16][hf * 16 + 8];
#pragma unroll
    for (int i = 0; i < 8; ++i) { ap[i] = plo[i]; ap[i + 8] = phi[i]; }

#pragma unroll
    for (int tn = 0; tn < 8; ++tn) {
      const bf16_t* vrow = vT + (long)(h * HDIM + tn * 16 + l16) * T_SEQ
                              + s0 + hf * 16;
      v16bf bv = *(const v16bf*)vrow;          // contiguous in k = s
      accy[tn] = __builtin_amdgcn_wmma_f32_16x16x32_bf16(false, ap, false, bv,
                                                         (short)0, accy[tn],
                                                         false, false);
    }
    __syncthreads();
  }

  // ---- normalize and store y as bf16 [T, H*D] ----
#pragma unroll
  for (int tn = 0; tn < 8; ++tn) {
#pragma unroll
    for (int r = 0; r < 8; ++r) {
      int trow = t0 + hf * 8 + r;
      float yv = accy[tn][r] / lrow[r];
      ybf[(long)trow * DIM + h * HDIM + tn * 16 + l16] = (bf16_t)yv;
    }
  }
}

// ---------------------------------------------------------------------------
extern "C" void kernel_launch(void* const* d_in, const int* in_sizes, int n_in,
                              void* d_out, int out_size, void* d_ws, size_t ws_size,
                              hipStream_t stream) {
  (void)in_sizes; (void)n_in; (void)out_size; (void)ws_size;
  const float* x    = (const float*)d_in[0];   // [T, DIM]
  const float* Wqkv = (const float*)d_in[1];   // [3*H*D, DIM]
  const float* Wout = (const float*)d_in[2];   // [DIM, H*D]
  float* out = (float*)d_out;                  // [T, DIM]

  char* ws = (char*)d_ws;
  size_t off = 0;
  bf16_t* xbf    = (bf16_t*)(ws + off); off += (size_t)T_SEQ * DIM * 2;      //  8 MB
  bf16_t* wqkvbf = (bf16_t*)(ws + off); off += (size_t)3 * DIM * DIM * 2;    // 24 MB
  bf16_t* woutbf = (bf16_t*)(ws + off); off += (size_t)DIM * DIM * 2;        //  8 MB
  float*  qkvf32 = (float*) (ws + off); off += (size_t)T_SEQ * FQKV * 4;     // 48 MB
  bf16_t* qbf    = (bf16_t*)(ws + off); off += (size_t)T_SEQ * DIM * 2;      //  8 MB
  bf16_t* kbf    = (bf16_t*)(ws + off); off += (size_t)T_SEQ * DIM * 2;      //  8 MB
  bf16_t* vT     = (bf16_t*)(ws + off); off += (size_t)T_SEQ * DIM * 2;      //  8 MB
  bf16_t* ybf    = (bf16_t*)(ws + off); off += (size_t)T_SEQ * DIM * 2;      //  8 MB

  // 1) convert inputs to bf16
  {
    int n = T_SEQ * DIM;
    f32_to_bf16_kernel<<<(n + 255) / 256, 256, 0, stream>>>(x, xbf, n);
    n = 3 * DIM * DIM;
    f32_to_bf16_kernel<<<(n + 255) / 256, 256, 0, stream>>>(Wqkv, wqkvbf, n);
    n = DIM * DIM;
    f32_to_bf16_kernel<<<(n + 255) / 256, 256, 0, stream>>>(Wout, woutbf, n);
  }

  // 2) QKV projection: qkv[T, FQKV] = x @ Wqkv^T
  gemm_bf16_wmma<<<dim3(FQKV / 128, T_SEQ / 16), 256, 0, stream>>>(
      xbf, wqkvbf, qkvf32, T_SEQ, FQKV, DIM);

  // 3) RMSNorm + rotary (q,k) and V transpose
  qk_prep_kernel<<<dim3(NHEADS, T_SEQ), 128, 0, stream>>>(qkvf32, qbf, kbf, vT);

  // 4) causal flash attention
  attn_kernel<<<dim3(T_SEQ / 16, NHEADS), 32, 0, stream>>>(qbf, kbf, vT, ybf);

  // 5) output projection: out[T, DIM] = y @ Wout^T
  gemm_bf16_wmma<<<dim3(DIM / 128, T_SEQ / 16), 256, 0, stream>>>(
      ybf, woutbf, out, T_SEQ, DIM, DIM);
}
